// FeatureNet_66743791780076
// MI455X (gfx1250) — matrix operand order repaired and checked by reading, not compiled, not run on hardware
//
#include <hip/hip_runtime.h>
#include <hip/hip_bf16.h>

typedef __attribute__((ext_vector_type(16))) _Float16 v16h;
typedef __attribute__((ext_vector_type(8)))  _Float16 v8h;
typedef __attribute__((ext_vector_type(8)))  float    v8f;

#define BB   8
#define NN   2048
#define KNB  8
#define DIM  128
#define NPTS (BB * NN)          // 16384 points
#define PTOT (NPTS * KNB)       // 131072 rows of activations
#define EPSB 1e-5f

// ---------------- ws layout (bytes) ----------------
// [0)                actA  f16 [PTOT][128]   33554432
// [33554432)         actB  f16 [PTOT][128]   33554432
// [67108864)         w1h   f16 [128][128]       32768
// [67141632)         w2h   f16 [128][128]       32768
// [67174400)         coef  f32 [3][2][128]       3072
#define OFF_ACTA 0
#define OFF_ACTB 33554432
#define OFF_W1H  67108864
#define OFF_W2H  67141632
#define OFF_COEF 67174400

// ---------------------------------------------------------------------------
// Kernel 1: convert weights to f16, fold conv-bias + eval-BN into (scale,shift)
// coef[l*256 + o]       = s_l[o] = g/sqrt(v+eps)
// coef[l*256 + 128 + o] = t_l[o] = s*b_conv + beta - m*s
// ---------------------------------------------------------------------------
__global__ __launch_bounds__(256) void prep_kernel(
    const float* __restrict__ ws2,   // [2,128,128]
    const float* __restrict__ b0,    // [128]
    const float* __restrict__ bs,    // [2,128]
    const float* __restrict__ bn_g, const float* __restrict__ bn_b,
    const float* __restrict__ bn_m, const float* __restrict__ bn_v,
    _Float16* __restrict__ w1h, _Float16* __restrict__ w2h,
    float* __restrict__ coef)
{
    int tid = blockIdx.x * blockDim.x + threadIdx.x;
    if (tid < 2 * DIM * DIM) {
        _Float16 h = (_Float16)ws2[tid];
        if (tid < DIM * DIM) w1h[tid] = h;
        else                 w2h[tid - DIM * DIM] = h;
    } else if (tid < 2 * DIM * DIM + 3 * DIM) {
        int i = tid - 2 * DIM * DIM;
        int l = i / DIM, o = i % DIM;
        float s = bn_g[l * DIM + o] * rsqrtf(bn_v[l * DIM + o] + EPSB);
        float bc = (l == 0) ? b0[o] : bs[(l - 1) * DIM + o];
        float t = s * bc + bn_b[l * DIM + o] - bn_m[l * DIM + o] * s;
        coef[l * 256 + o]       = s;
        coef[l * 256 + 128 + o] = t;
    }
}

// ---------------------------------------------------------------------------
// Kernel 2: one wave32 per point. kNN top-8 + relative coords + block0
// (3->128 conv + BN + ReLU), output f16 activations [PTOT][128].
// ---------------------------------------------------------------------------
__global__ __launch_bounds__(256) void knn_block0_kernel(
    const float* __restrict__ x,     // [B,3,N]
    const float* __restrict__ w0,    // [128,3]
    const float* __restrict__ coef,  // layer-0 scale/shift at coef[0..255]
    _Float16* __restrict__ act)      // [PTOT][128]
{
    const int lane = threadIdx.x & 31;
    const int pt   = blockIdx.x * 8 + (threadIdx.x >> 5);   // 0..16383
    const int b    = pt >> 11;
    const int n    = pt & 2047;
    const float* xb = x + (size_t)b * 3 * NN;

    const float xn0 = xb[n], xn1 = xb[NN + n], xn2 = xb[2 * NN + n];

    const float FINF = 3.402823466e38f;
    float bd[KNB]; int bi[KNB];
#pragma unroll
    for (int j = 0; j < KNB; ++j) { bd[j] = FINF; bi[j] = 0; }

    // strided scan: rank by |xm|^2 - 2*xn.xm  (== dist - |xn|^2; self is global min)
    for (int m = lane; m < NN; m += 32) {
        float m0 = xb[m], m1 = xb[NN + m], m2 = xb[2 * NN + m];
        float d = m0 * m0 + m1 * m1 + m2 * m2
                - 2.0f * (xn0 * m0 + xn1 * m1 + xn2 * m2);
        float cd = d; int ci = m;
#pragma unroll
        for (int j = 0; j < KNB; ++j) {   // branchless sorted insertion
            bool lt = cd < bd[j];
            float td = bd[j]; int ti = bi[j];
            bd[j] = lt ? cd : td;  bi[j] = lt ? ci : ti;
            cd    = lt ? td : cd;  ci    = lt ? ti : ci;
        }
    }

    // merge: 8 rounds of wave-argmin (wave32 butterfly), winner pops its head
    int nbr[KNB];
#pragma unroll
    for (int r = 0; r < KNB; ++r) {
        float v = bd[0]; int who = lane;
#pragma unroll
        for (int off = 16; off >= 1; off >>= 1) {
            float ov = __shfl_xor(v, off, 32);
            int   ow = __shfl_xor(who, off, 32);
            if (ov < v || (ov == v && ow < who)) { v = ov; who = ow; }
        }
        nbr[r] = __shfl(bi[0], who, 32);
        if (lane == who) {
#pragma unroll
            for (int j = 0; j < KNB - 1; ++j) { bd[j] = bd[j + 1]; bi[j] = bi[j + 1]; }
            bd[KNB - 1] = FINF;
        }
    }

    // block0: each lane handles channels o = lane + 32*j, j = 0..3
    float wA[4], wB[4], wC[4], sc[4], sh[4];
#pragma unroll
    for (int j = 0; j < 4; ++j) {
        int o = lane + 32 * j;
        wA[j] = w0[o * 3 + 0]; wB[j] = w0[o * 3 + 1]; wC[j] = w0[o * 3 + 2];
        sc[j] = coef[o]; sh[j] = coef[128 + o];
    }
#pragma unroll
    for (int k = 0; k < KNB; ++k) {
        int mi = nbr[k];
        float d0 = xb[mi] - xn0;
        float d1 = xb[NN + mi] - xn1;
        float d2 = xb[2 * NN + mi] - xn2;
        _Float16* dst = act + ((size_t)pt * KNB + k) * DIM;
#pragma unroll
        for (int j = 0; j < 4; ++j) {
            float z = fmaf(wA[j], d0, fmaf(wB[j], d1, wC[j] * d2));
            float y = fmaxf(fmaf(z, sc[j], sh[j]), 0.0f);
            dst[lane + 32 * j] = (_Float16)y;
        }
    }
}

// ---------------------------------------------------------------------------
// Kernel 3: WMMA GEMM  y[p,o] = ReLU(s[o] * (sum_c W[o,c]*act[p,c]) + t[o])
// A = act tile (16 pts x 32 ch), B = W rows (32 ch x 16 out), f32 accum.
// out_act != nullptr: store f16 activations.
// out_max != nullptr: per-lane max over its 8 D-rows (= one K-group) -> [B,128,N]
// ---------------------------------------------------------------------------
__global__ __launch_bounds__(256) void gemm_wmma_kernel(
    const _Float16* __restrict__ act,  // [PTOT][128]
    const _Float16* __restrict__ w,    // [128][128] row = out channel
    const float*    __restrict__ coef, // s[0..127], t[128..255]
    _Float16* __restrict__ out_act,    // nullable
    float*    __restrict__ out_max)    // nullable
{
    const int lane = threadIdx.x & 31;
    const int m    = lane & 15;        // A row / B column within tile
    const int hi   = lane >> 4;        // half-wave selector
    const int tile = blockIdx.x * 8 + (threadIdx.x >> 5);   // 8192 tiles
    const int pbase = tile * 16;

    // ---- load 4 A fragments (16 points x full K=128), contiguous k = 16*hi + s ----
    const _Float16* arow = act + (size_t)(pbase + m) * DIM;
    v16h a[4];
#pragma unroll
    for (int ks = 0; ks < 4; ++ks) {
        const int k0 = ks * 32;
        v8h lo = *(const v8h*)(arow + k0 + hi * 16);
        v8h hv = *(const v8h*)(arow + k0 + hi * 16 + 8);
        a[ks] = __builtin_shufflevector(lo, hv,
                 0,1,2,3,4,5,6,7,8,9,10,11,12,13,14,15);
    }

#pragma unroll
    for (int ot = 0; ot < 8; ++ot) {
        const int obase = ot * 16;
        const int o = obase + m;
        v8f acc = {};
#pragma unroll
        for (int ks = 0; ks < 4; ++ks) {
            // B lane m holds W[obase+m][k0 + hi*16 .. +15] (contiguous 32B)
            v16h bf = *(const v16h*)(w + (size_t)o * DIM + ks * 32 + hi * 16);
            acc = __builtin_amdgcn_wmma_f32_16x16x32_f16(
                      false, a[ks], false, bf, (short)0, acc, false, false);
        }
        const float s = coef[o], t = coef[128 + o];
        if (out_max) {
            // lane's 8 rows = points pbase + hi*8 + r = one full K-group
            float mv = 0.0f;
#pragma unroll
            for (int r = 0; r < 8; ++r)
                mv = fmaxf(mv, fmaxf(fmaf(acc[r], s, t), 0.0f));
            const int g = tile * 2 + hi;          // point id 0..16383
            const int b = g >> 11, n = g & 2047;
            out_max[((size_t)b * DIM + o) * NN + n] = mv;
        } else {
#pragma unroll
            for (int r = 0; r < 8; ++r) {
                float y = fmaxf(fmaf(acc[r], s, t), 0.0f);
                out_act[(size_t)(pbase + hi * 8 + r) * DIM + o] = (_Float16)y;
            }
        }
    }
}

// ---------------------------------------------------------------------------
extern "C" void kernel_launch(void* const* d_in, const int* in_sizes, int n_in,
                              void* d_out, int out_size, void* d_ws, size_t ws_size,
                              hipStream_t stream) {
    const float* x    = (const float*)d_in[0];
    const float* w0   = (const float*)d_in[1];
    const float* b0   = (const float*)d_in[2];
    const float* ws2  = (const float*)d_in[3];
    const float* bs   = (const float*)d_in[4];
    const float* bn_g = (const float*)d_in[5];
    const float* bn_b = (const float*)d_in[6];
    const float* bn_m = (const float*)d_in[7];
    const float* bn_v = (const float*)d_in[8];

    char* wsb = (char*)d_ws;
    _Float16* actA = (_Float16*)(wsb + OFF_ACTA);
    _Float16* actB = (_Float16*)(wsb + OFF_ACTB);
    _Float16* w1h  = (_Float16*)(wsb + OFF_W1H);
    _Float16* w2h  = (_Float16*)(wsb + OFF_W2H);
    float*    coef = (float*)   (wsb + OFF_COEF);
    float*    out  = (float*)d_out;

    // 1) weight f16 conversion + BN folding
    prep_kernel<<<(2 * DIM * DIM + 3 * DIM + 255) / 256, 256, 0, stream>>>(
        ws2, b0, bs, bn_g, bn_b, bn_m, bn_v, w1h, w2h, coef);

    // 2) kNN + grouping + block0 (3->128): 16384 waves
    knn_block0_kernel<<<NPTS / 8, 256, 0, stream>>>(x, w0, coef, actA);

    // 3) block1 (128->128) WMMA, f16 out
    gemm_wmma_kernel<<<(PTOT / 16) / 8, 256, 0, stream>>>(
        actA, w1h, coef + 256, actB, nullptr);

    // 4) block2 (128->128) WMMA + fused max-over-K, f32 out [B,128,N]
    gemm_wmma_kernel<<<(PTOT / 16) / 8, 256, 0, stream>>>(
        actB, w2h, coef + 512, nullptr, out);
}
